// MaskedAttention_38259568673144
// MI455X (gfx1250) — matrix operand order repaired and checked by reading, not compiled, not run on hardware
//
#include <hip/hip_runtime.h>

// Masked (causal) multi-head attention, GPT-2 style.
// B=4, S=2048, NX=1024, NH=16, HD=64. All GEMMs run on v_wmma_f32_16x16x32_bf16
// (fp32 accumulate). Dense-GEMM B operands staged into LDS by the Tensor Data
// Mover (tensor_load_to_lds, TENSORcnt-tracked, double buffered).

#define BB  4
#define SSQ 2048
#define NXX 1024
#define NHH 16
#define HDD 64

// LDS row stride for TDM-staged B tiles: 32 bf16 data + 16 bf16 pad = 96 bytes
#define BLD 48

typedef __attribute__((ext_vector_type(16))) __bf16          v16bf;
typedef __attribute__((ext_vector_type(8)))  float           v8f;
typedef __attribute__((ext_vector_type(8)))  unsigned short  ushort8;
typedef __attribute__((ext_vector_type(16))) unsigned short  ushort16;
typedef __attribute__((ext_vector_type(4)))  unsigned int    v4u;
typedef __attribute__((ext_vector_type(8)))  int             v8i;
typedef __attribute__((ext_vector_type(4)))  int             v4i;

static __device__ __forceinline__ unsigned short f2bf(float f) {
  unsigned u = __builtin_bit_cast(unsigned, f);
  unsigned r = u + 0x7FFFu + ((u >> 16) & 1u);   // round-to-nearest-even
  return (unsigned short)(r >> 16);
}

// A-operand tile (16 rows x 32 K, bf16) from a row-major [*, ld] bf16 array.
// Lane L holds row row0 + (L&15); K elements: [kb+8g .. kb+8g+7] then
// [kb+16+8g .. kb+16+8g+7] per the 16-bit A layout (g = L>>4).
static __device__ __forceinline__ v16bf load_a_tile(const unsigned short* p,
                                                    int row0, int ld, int kb) {
  int lane = threadIdx.x & 31;
  int r = row0 + (lane & 15);
  int g = lane >> 4;
  const ushort8* p0 = (const ushort8*)(p + (size_t)r * ld + kb + 8 * g);
  const ushort8* p1 = (const ushort8*)(p + (size_t)r * ld + kb + 16 + 8 * g);
  ushort16 t;
  t.lo = *p0;
  t.hi = *p1;
  return __builtin_bit_cast(v16bf, t);
}

// B-operand tile (32 K x 16 cols, bf16). Lane L holds column col0 + (L&15);
// K elements: [kb+8g .. kb+8g+7] then [kb+16+8g .. kb+16+8g+7]. Source must be stored so that
// B's columns are K-contiguous rows (transposed weight / k rows / vT rows).
static __device__ __forceinline__ v16bf load_b_tile(const unsigned short* p,
                                                    int col0, int ld, int kb) {
  int lane = threadIdx.x & 31;
  int r = col0 + (lane & 15);
  int g = lane >> 4;
  const ushort8* p0 = (const ushort8*)(p + (size_t)r * ld + kb + 8 * g);
  const ushort8* p1 = (const ushort8*)(p + (size_t)r * ld + kb + 16 + 8 * g);
  ushort16 t;
  t.lo = *p0;
  t.hi = *p1;
  return __builtin_bit_cast(v16bf, t);
}

#define WMMA_BF16(a, b, c) \
  __builtin_amdgcn_wmma_f32_16x16x32_bf16(false, (a), false, (b), (short)0, (c), false, false)

// Issue one TDM 2D tile load: tile_d0 x tile_d1 bf16 elements from a row-major
// [*, stride_elems] bf16 tensor into LDS at byte offset lds_off, with LDS
// padding of 8 DWORDs every 16 DWORDs (row stride 96B = BLD bf16).
// D# packing per CDNA5 ISA ch.8 (group0: count/lds/global/type; group1: dims).
// clang-23 toolchain: 6-arg builtin (g0 v4u, g1 v8i, v4i, v4i, v8i, i32 cpol).
static __device__ __forceinline__ void tdm_load_2d(unsigned lds_off,
                                                   const unsigned short* gptr,
                                                   unsigned tensor_d0,
                                                   unsigned tensor_d1,
                                                   unsigned stride_elems,
                                                   unsigned tile_d0,
                                                   unsigned tile_d1) {
  unsigned long long ga = (unsigned long long)(size_t)gptr;
  v4u g0;
  g0[0] = 1u;                                            // count=1, user mode
  g0[1] = lds_off;                                       // lds_addr (bytes)
  g0[2] = (unsigned)(ga & 0xFFFFFFFFu);                  // global_addr[31:0]
  g0[3] = (unsigned)((ga >> 32) & 0x01FFFFFFu) | (2u << 30); // [56:32] | type=2
  v8i g1;
  // data_size=1 (2B) | pad_enable | pad_interval=3 (16 DW) | pad_amount=7 (8 DW)
  g1[0] = (int)((1u << 16) | (1u << 20) | (3u << 22) | (7u << 25));
  g1[1] = (int)((tensor_d0 & 0xFFFFu) << 16);            // tensor_dim0[15:0]
  g1[2] = (int)(((tensor_d0 >> 16) & 0xFFFFu) | ((tensor_d1 & 0xFFFFu) << 16));
  g1[3] = (int)(((tensor_d1 >> 16) & 0xFFFFu) | ((tile_d0 & 0xFFFFu) << 16));
  g1[4] = (int)(tile_d1 & 0xFFFFu);                      // tile_dim1 (tile_dim2=0)
  g1[5] = (int)stride_elems;                             // tensor_dim0_stride[31:0]
  g1[6] = 0;                                             // stride[47:32], dim1_stride lo
  g1[7] = 0;
  v4i z4 = {};
  v8i z8 = {};
  __builtin_amdgcn_tensor_load_to_lds(g0, g1, z4, z4, z8, 0);
}

// ---------------------------------------------------------------- converters
__global__ void cvt_bf16_kernel(const float* __restrict__ in,
                                unsigned short* __restrict__ out, int n) {
  int i = blockIdx.x * blockDim.x + threadIdx.x;
  if (i < n) out[i] = f2bf(in[i]);
}

// out[n][k] = bf16(in[k][n]); in is [rowsK, colsN] row-major.
__global__ void transpose_bf16_kernel(const float* __restrict__ in,
                                      unsigned short* __restrict__ out,
                                      int rowsK, int colsN) {
  int i = blockIdx.x * blockDim.x + threadIdx.x;
  if (i >= rowsK * colsN) return;
  int n = i / rowsK;
  int k = i % rowsK;
  out[(size_t)n * rowsK + k] = f2bf(in[(size_t)k * colsN + n]);
}

// --------------------------------------------------------------- QKV GEMM
// y = x @ W + b, y split into q,k ([B,NH,S,HD] bf16) and v transposed
// ([B,NH,HD,S] bf16). Wave computes 16x64; block = 8 waves = 128x64 tile.
// B super-tile (64 rows x 32 K) TDM-staged into LDS, double buffered.
__global__ __launch_bounds__(256) void qkv_gemm_kernel(
    const unsigned short* __restrict__ xbf,    // [8192,1024] bf16
    const unsigned short* __restrict__ wT,     // [3072,1024] bf16 (W^T)
    const float* __restrict__ bias,            // [3072]
    unsigned short* __restrict__ qb,
    unsigned short* __restrict__ kb_,
    unsigned short* __restrict__ vtb) {
  __shared__ unsigned short bsm[2][64 * BLD];  // 2 x 6 KiB TDM destination
  int wave = threadIdx.x >> 5;
  int lane = threadIdx.x & 31;
  int m0 = (blockIdx.y * 8 + wave) * 16;       // 0..8191
  int n0 = blockIdx.x * 64;                    // 0..3071

  if (wave == 0)
    tdm_load_2d((unsigned)(size_t)&bsm[0][0], wT + (size_t)n0 * NXX,
                NXX, 3 * NXX, NXX, 32, 64);

  v8f acc[4] = {};
  for (int kb = 0; kb < NXX; kb += 32) {
    int cur = (kb >> 5) & 1;
    if (wave == 0) {
      if (kb + 32 < NXX) {
        tdm_load_2d((unsigned)(size_t)&bsm[cur ^ 1][0],
                    wT + (size_t)n0 * NXX + kb + 32, NXX, 3 * NXX, NXX, 32, 64);
        __builtin_amdgcn_s_wait_tensorcnt(1);
      } else {
        __builtin_amdgcn_s_wait_tensorcnt(0);
      }
    }
    __syncthreads();                           // bsm[cur] ready
    if (kb + 64 < NXX)
      __builtin_prefetch(xbf + (size_t)(m0 + (lane & 15)) * NXX + kb + 64, 0, 0);
    v16bf a = load_a_tile(xbf, m0, NXX, kb);
#pragma unroll
    for (int t = 0; t < 4; ++t) {
      v16bf b = load_b_tile(&bsm[cur][0], 16 * t, BLD, 0);
      acc[t] = WMMA_BF16(a, b, acc[t]);
    }
    __syncthreads();                           // all done reading bsm[cur]
  }

  int g = lane >> 4;
  int nc = lane & 15;
#pragma unroll
  for (int t = 0; t < 4; ++t) {
    int n = n0 + 16 * t + nc;
    int sec = n / NXX;                         // 0=q 1=k 2=v
    int nn = n % NXX;
    int h = nn / HDD;
    int d = nn % HDD;
    float bv = bias[n];
#pragma unroll
    for (int e = 0; e < 8; ++e) {
      int m = m0 + e + 8 * g;
      int bI = m / SSQ;
      int s = m % SSQ;
      unsigned short val = f2bf(acc[t][e] + bv);
      size_t bh = (size_t)bI * NHH + h;
      if (sec == 0)      qb [(bh * SSQ + s) * HDD + d] = val;
      else if (sec == 1) kb_[(bh * SSQ + s) * HDD + d] = val;
      else               vtb[(bh * HDD + d) * SSQ + s] = val;   // store v^T
    }
  }
}

// ------------------------------------------------------- flash attention
// One wave per 16-query tile; causal loop over 32-key chunks; online softmax.
__global__ __launch_bounds__(128) void attn_kernel(
    const unsigned short* __restrict__ qb,     // [B*NH, S, HD] bf16
    const unsigned short* __restrict__ kbuf,   // [B*NH, S, HD] bf16
    const unsigned short* __restrict__ vtb,    // [B*NH, HD, S] bf16
    unsigned short* __restrict__ abuf) {       // [B, S, NX]   bf16
  __shared__ unsigned short plds[4][16 * 32];  // per-wave prob transpose tile
  int wave = threadIdx.x >> 5;
  int lane = threadIdx.x & 31;
  int g = lane >> 4;
  int nc = lane & 15;
  int bh = blockIdx.x;
  int bI = bh / NHH;
  int h  = bh % NHH;
  int q0 = (blockIdx.y * 4 + wave) * 16;

  const unsigned short* qp = qb   + (size_t)bh * SSQ * HDD;
  const unsigned short* kp = kbuf + (size_t)bh * SSQ * HDD;
  const unsigned short* vp = vtb  + (size_t)bh * HDD * SSQ;

  v16bf aq0 = load_a_tile(qp, q0, HDD, 0);     // q rows, d 0..31
  v16bf aq1 = load_a_tile(qp, q0, HDD, 32);    // q rows, d 32..63

  v8f o[4] = {};
  float mrow[8], lrow[8];
#pragma unroll
  for (int e = 0; e < 8; ++e) { mrow[e] = -3.0e38f; lrow[e] = 0.0f; }

  for (int kb = 0; kb <= q0 + 15; kb += 32) {
    // scores: keys [kb,kb+16) and [kb+16,kb+32), K-depth = HD = 64
    v8f s0 = {};
    v8f s1 = {};
    s0 = WMMA_BF16(aq0, load_b_tile(kp, kb,      HDD, 0),  s0);
    s0 = WMMA_BF16(aq1, load_b_tile(kp, kb,      HDD, 32), s0);
    s1 = WMMA_BF16(aq0, load_b_tile(kp, kb + 16, HDD, 0),  s1);
    s1 = WMMA_BF16(aq1, load_b_tile(kp, kb + 16, HDD, 32), s1);

    float corr[8];
#pragma unroll
    for (int e = 0; e < 8; ++e) {
      int q  = q0 + e + 8 * g;
      float v0 = (kb + nc      <= q) ? s0[e] * 0.125f : -10000.0f;
      float v1 = (kb + 16 + nc <= q) ? s1[e] * 0.125f : -10000.0f;
      float rm = fmaxf(v0, v1);
#pragma unroll
      for (int off = 1; off < 16; off <<= 1)
        rm = fmaxf(rm, __shfl_xor(rm, off, 32));
      float mn = fmaxf(mrow[e], rm);
      float c  = __expf(mrow[e] - mn);
      float p0 = __expf(v0 - mn);
      float p1 = __expf(v1 - mn);
      float rs = p0 + p1;
#pragma unroll
      for (int off = 1; off < 16; off <<= 1)
        rs += __shfl_xor(rs, off, 32);
      lrow[e] = lrow[e] * c + rs;
      mrow[e] = mn;
      corr[e] = c;
      int row = e + 8 * g;
      plds[wave][row * 32 + nc]      = f2bf(p0);
      plds[wave][row * 32 + 16 + nc] = f2bf(p1);
    }
#pragma unroll
    for (int t = 0; t < 4; ++t)
#pragma unroll
      for (int e = 0; e < 8; ++e)
        o[t][e] *= corr[e];

    // probs 16x32 (A layout via LDS round-trip), V^T rows K-contiguous
    v16bf ap = load_a_tile(&plds[wave][0], 0, 32, 0);
#pragma unroll
    for (int t = 0; t < 4; ++t) {
      v16bf bv = load_b_tile(vp, t * 16, SSQ, kb);
      o[t] = WMMA_BF16(ap, bv, o[t]);
    }
  }

#pragma unroll
  for (int t = 0; t < 4; ++t) {
#pragma unroll
    for (int e = 0; e < 8; ++e) {
      int s = q0 + e + 8 * g;
      int col = h * HDD + t * 16 + nc;
      abuf[((size_t)bI * SSQ + s) * NXX + col] = f2bf(o[t][e] / lrow[e]);
    }
  }
}

// --------------------------------------------------------------- proj GEMM
__global__ __launch_bounds__(256) void proj_gemm_kernel(
    const unsigned short* __restrict__ abuf,   // [8192,1024] bf16
    const unsigned short* __restrict__ wpT,    // [1024,1024] bf16 (W^T)
    const float* __restrict__ bias,            // [1024]
    float* __restrict__ out) {                 // [8192,1024] f32
  __shared__ unsigned short bsm[2][64 * BLD];
  int wave = threadIdx.x >> 5;
  int lane = threadIdx.x & 31;
  int m0 = (blockIdx.y * 8 + wave) * 16;
  int n0 = blockIdx.x * 64;

  if (wave == 0)
    tdm_load_2d((unsigned)(size_t)&bsm[0][0], wpT + (size_t)n0 * NXX,
                NXX, NXX, NXX, 32, 64);

  v8f acc[4] = {};
  for (int kb = 0; kb < NXX; kb += 32) {
    int cur = (kb >> 5) & 1;
    if (wave == 0) {
      if (kb + 32 < NXX) {
        tdm_load_2d((unsigned)(size_t)&bsm[cur ^ 1][0],
                    wpT + (size_t)n0 * NXX + kb + 32, NXX, NXX, NXX, 32, 64);
        __builtin_amdgcn_s_wait_tensorcnt(1);
      } else {
        __builtin_amdgcn_s_wait_tensorcnt(0);
      }
    }
    __syncthreads();
    if (kb + 64 < NXX)
      __builtin_prefetch(abuf + (size_t)(m0 + (lane & 15)) * NXX + kb + 64, 0, 0);
    v16bf a = load_a_tile(abuf, m0, NXX, kb);
#pragma unroll
    for (int t = 0; t < 4; ++t) {
      v16bf b = load_b_tile(&bsm[cur][0], 16 * t, BLD, 0);
      acc[t] = WMMA_BF16(a, b, acc[t]);
    }
    __syncthreads();
  }

  int g = lane >> 4;
  int nc = lane & 15;
#pragma unroll
  for (int t = 0; t < 4; ++t) {
    int n = n0 + 16 * t + nc;
    float bv = bias[n];
#pragma unroll
    for (int e = 0; e < 8; ++e) {
      int m = m0 + e + 8 * g;
      out[(size_t)m * NXX + n] = acc[t][e] + bv;
    }
  }
}

// ----------------------------------------------------------------- launcher
extern "C" void kernel_launch(void* const* d_in, const int* in_sizes, int n_in,
                              void* d_out, int out_size, void* d_ws, size_t ws_size,
                              hipStream_t stream) {
  const float* x      = (const float*)d_in[0];   // [4,2048,1024]
  const float* w_attn = (const float*)d_in[1];   // [1024,3072]
  const float* b_attn = (const float*)d_in[2];   // [3072]
  const float* w_proj = (const float*)d_in[3];   // [1024,1024]
  const float* b_proj = (const float*)d_in[4];   // [1024]
  float* out = (float*)d_out;

  char* ws = (char*)d_ws;
  const size_t MiB = 1024 * 1024;
  unsigned short* xbf    = (unsigned short*)(ws + 0 * MiB);   // 16 MiB
  unsigned short* wattnT = (unsigned short*)(ws + 16 * MiB);  //  6 MiB
  unsigned short* wprojT = (unsigned short*)(ws + 22 * MiB);  //  2 MiB
  unsigned short* qb     = (unsigned short*)(ws + 24 * MiB);  // 16 MiB
  unsigned short* kb     = (unsigned short*)(ws + 40 * MiB);  // 16 MiB
  unsigned short* vtb    = (unsigned short*)(ws + 56 * MiB);  // 16 MiB
  unsigned short* abuf   = (unsigned short*)(ws + 72 * MiB);  // 16 MiB (total 88)

  int nx = BB * SSQ * NXX;
  cvt_bf16_kernel<<<(nx + 255) / 256, 256, 0, stream>>>(x, xbf, nx);
  transpose_bf16_kernel<<<(NXX * 3 * NXX + 255) / 256, 256, 0, stream>>>(
      w_attn, wattnT, NXX, 3 * NXX);
  transpose_bf16_kernel<<<(NXX * NXX + 255) / 256, 256, 0, stream>>>(
      w_proj, wprojT, NXX, NXX);

  qkv_gemm_kernel<<<dim3(48, 64), 256, 0, stream>>>(xbf, wattnT, b_attn, qb, kb, vtb);
  attn_kernel<<<dim3(BB * NHH, SSQ / 16 / 4), 128, 0, stream>>>(qb, kb, vtb, abuf);
  proj_gemm_kernel<<<dim3(16, 64), 256, 0, stream>>>(abuf, wprojT, b_proj, out);
}
